// GCN_one_pyg_86758339379593
// MI455X (gfx1250) — matrix operand order repaired and checked by reading, not compiled, not run on hardware
//
#include <hip/hip_runtime.h>

// GCN layer: out = relu(D^-1/2 A_hat D^-1/2 (feat @ W) + conv_bias + extra_bias)
// N=10000, IN_FT=512, OUT_FT=256. A_hat = adj with diagonal forced to 1.0.
//
// MI455X plan: adjacency (400MB fp32) streams from HBM twice (degree pass +
// aggregation pass) -> ~34us floor at 23.3 TB/s. The aggregation has
// 128 flop/byte intensity, so it must run on v_wmma_f32_16x16x32_bf16.
// Each wave owns a 16x256 output strip (16 f32 accumulators, 128 VGPRs):
// every A element is loaded & converted to bf16 exactly once and feeds 16
// WMMAs. B (dinv-scaled h, 5.1MB bf16, K-major) stays L2-resident; its 16
// tiles are reached via compile-time instruction offsets off one base ptr.

#define NN      10000
#define INF     512
#define OUTF    256
#define KPAD    10016   // NN rounded up to a multiple of 32
#define NSTRIP  (NN / 16)   // 625 row strips

typedef __attribute__((ext_vector_type(16))) __bf16          v16bf;
typedef __attribute__((ext_vector_type(8)))  float           v8f;
typedef __attribute__((ext_vector_type(8)))  unsigned int    v8u;
typedef __attribute__((ext_vector_type(4)))  float           v4f;

__device__ __forceinline__ v16bf load16bf(const __bf16* p) {
  v8u u = *(const v8u*)p;                   // two global_load_b128
  return __builtin_bit_cast(v16bf, u);
}

__device__ __forceinline__ v8f wmma_bf16(v16bf a, v16bf b, v8f c) {
  return __builtin_amdgcn_wmma_f32_16x16x32_bf16(
      /*neg_a=*/false, a, /*neg_b=*/false, b,
      /*c_mod=*/(short)0, c, /*reuse_a=*/false, /*reuse_b=*/false);
}

// ---------------------------------------------------------------- kernel 1
// dinv[i] = rsqrt(sum_j A_hat[i][j]) ; A_hat diag = 1.0
__global__ void deg_kernel(const float* __restrict__ adj,
                           float* __restrict__ dinv) {
  __shared__ float red[256];
  const int row = blockIdx.x;
  const float* r = adj + (size_t)row * NN;
  float s = 0.f;
  const v4f* r4 = (const v4f*)r;
  for (int i = threadIdx.x; i < NN / 4; i += 256) {
    v4f v = r4[i];
    s += v[0] + v[1] + v[2] + v[3];
  }
  red[threadIdx.x] = s;
  __syncthreads();
  for (int off = 128; off > 0; off >>= 1) {
    if (threadIdx.x < off) red[threadIdx.x] += red[threadIdx.x + off];
    __syncthreads();
  }
  if (threadIdx.x == 0) {
    float deg = red[0] - r[row] + 1.0f;     // replace diag with 1.0
    dinv[row] = deg > 0.f ? rsqrtf(deg) : 0.f;
  }
}

// ---------------------------------------------------------------- kernel 2
// WT[f][k] = bf16(W[k][f]); also zero hsT K-padding (k in [NN, KPAD)).
__global__ void wt_kernel(const float* __restrict__ W,
                          __bf16* __restrict__ WT,
                          __bf16* __restrict__ hsT) {
  const int idx = blockIdx.x * 256 + threadIdx.x;
  if (idx < INF * OUTF) {
    const int f = idx >> 9, k = idx & (INF - 1);
    WT[idx] = (__bf16)W[k * OUTF + f];
  }
  if (idx < OUTF * (KPAD - NN)) {
    const int f = idx >> 4, k = NN + (idx & 15);
    hsT[(size_t)f * KPAD + k] = (__bf16)0.0f;
  }
}

// ---------------------------------------------------------------- kernel 3
// hsT[f][j] = bf16(dinv[j] * (feat @ W)[j][f])   (WMMA bf16, K=512)
// Block: 256 thr = 8 waves, 16 rows x 256 cols. Wave w: cols [w*32, w*32+32).
__global__ void h_kernel(const float* __restrict__ feat,
                         const __bf16* __restrict__ WT,
                         const float* __restrict__ dinv,
                         __bf16* __restrict__ hsT) {
  const int w = threadIdx.x >> 5, lane = threadIdx.x & 31;
  const int half = lane >> 4, l16 = lane & 15;
  const int J0 = blockIdx.x * 16;
  const int F0 = w * 32;
  v8f c0 = {}, c1 = {};
  const float* arow = feat + (size_t)(J0 + l16) * INF;
  const __bf16* b0p = WT + (size_t)(F0 + l16) * INF;
  const __bf16* b1p = WT + (size_t)(F0 + 16 + l16) * INF;

  for (int k0 = 0; k0 < INF; k0 += 32) {
    // A tile: lane holds row J0+l16, k = e + half*16 + k0
    const float* p1 = arow + k0 + half * 16;
    v4f x0 = *(const v4f*)(p1);
    v4f x1 = *(const v4f*)(p1 + 4);
    v4f x2 = *(const v4f*)(p1 + 8);
    v4f x3 = *(const v4f*)(p1 + 12);
    v16bf a;
#pragma unroll
    for (int e = 0; e < 4; ++e) {
      a[e]      = (__bf16)x0[e];
      a[e + 4]  = (__bf16)x1[e];
      a[e + 8]  = (__bf16)x2[e];
      a[e + 12] = (__bf16)x3[e];
    }
    // B tiles: lane holds col, k = e + half*16 + k0 (contiguous in WT)
    const int kb = k0 + half * 16;
    v16bf b0 = load16bf(b0p + kb);
    v16bf b1 = load16bf(b1p + kb);
    c0 = wmma_bf16(a, b0, c0);
    c1 = wmma_bf16(a, b1, c1);
  }
  // D layout: c[v] -> row J0 + v + half*8, col F0 + l16 (+16 for c1)
#pragma unroll
  for (int v = 0; v < 8; ++v) {
    const int j = J0 + v + half * 8;
    const float dv = dinv[j];
    hsT[(size_t)(F0 + l16) * KPAD + j]      = (__bf16)(dv * c0[v]);
    hsT[(size_t)(F0 + 16 + l16) * KPAD + j] = (__bf16)(dv * c1[v]);
  }
}

// ---------------------------------------------------------------- kernel 4
// out = relu(dinv[i] * (A_hat_bf16 @ hs) + conv_bias + extra_bias)
// One wave per 16-row strip, covering ALL 256 output cols (16 accumulators).
// A loaded/converted once per element; 16 WMMAs per K-step of 32.
__global__ void agg_kernel(const float* __restrict__ adj,
                           const __bf16* __restrict__ hsT,
                           const float* __restrict__ dinv,
                           const float* __restrict__ cb,
                           const float* __restrict__ eb,
                           float* __restrict__ out) {
  const int w = threadIdx.x >> 5, lane = threadIdx.x & 31;
  const int half = lane >> 4, l16 = lane & 15;
  const int strip = blockIdx.x * 8 + w;
  if (strip >= NSTRIP) return;               // wave-uniform tail guard
  const int R0 = strip * 16;
  const int m = R0 + l16;                    // this lane's A row

  v8f c[16];
#pragma unroll
  for (int t = 0; t < 16; ++t) c[t] = (v8f){};

  const float* arow = adj + (size_t)m * NN;
  // lane's B column base for tile 0; tile t adds t*16*KPAD (const offset)
  const __bf16* bbase = hsT + (size_t)l16 * KPAD + half * 8;

  for (int k0 = 0; k0 < NN; k0 += 32) {
    const int korig1 = k0 + half * 8;
    const int korig2 = korig1 + 16;
    const int kb2 = (korig2 <= NN - 8) ? korig2 : (NN - 8);  // tail clamp
    v4f x0 = *(const v4f*)(arow + korig1);
    v4f x1 = *(const v4f*)(arow + korig1 + 4);
    v4f x2 = *(const v4f*)(arow + kb2);
    v4f x3 = *(const v4f*)(arow + kb2 + 4);
    // (clamped tail values multiply zero-padded B, so they are harmless)

    v16bf a;
    const bool diagWindow = (R0 <= k0 + 31) && (k0 <= R0 + 15); // uniform
    if (diagWindow) {
#pragma unroll
      for (int e = 0; e < 4; ++e) {
        a[e]      = (__bf16)((korig1 + e     == m) ? 1.0f : x0[e]);
        a[e + 4]  = (__bf16)((korig1 + e + 4 == m) ? 1.0f : x1[e]);
        a[e + 8]  = (__bf16)((korig2 + e     == m) ? 1.0f : x2[e]);
        a[e + 12] = (__bf16)((korig2 + e + 4 == m) ? 1.0f : x3[e]);
      }
    } else {
#pragma unroll
      for (int e = 0; e < 4; ++e) {
        a[e]      = (__bf16)x0[e];
        a[e + 4]  = (__bf16)x1[e];
        a[e + 8]  = (__bf16)x2[e];
        a[e + 12] = (__bf16)x3[e];
      }
    }

#pragma unroll
    for (int t = 0; t < 16; ++t) {
      const __bf16* bp = bbase + (size_t)t * 16 * KPAD + k0;
      v16bf b;
#pragma unroll
      for (int s = 0; s < 8; ++s) { b[s] = bp[s]; b[s + 8] = bp[s + 16]; }
      c[t] = wmma_bf16(a, b, c[t]);
    }
  }

#pragma unroll
  for (int v = 0; v < 8; ++v) {
    const int row = R0 + v + half * 8;
    const float scale = dinv[row];
#pragma unroll
    for (int t = 0; t < 16; ++t) {
      const int f = t * 16 + l16;
      float o = scale * c[t][v] + cb[f] + eb[f];
      out[(size_t)row * OUTF + f] = o > 0.f ? o : 0.f;
    }
  }
}

// ----------------------------------------------------------------
extern "C" void kernel_launch(void* const* d_in, const int* in_sizes, int n_in,
                              void* d_out, int out_size, void* d_ws, size_t ws_size,
                              hipStream_t stream) {
  const float* feat = (const float*)d_in[0];   // [N, IN_FT]
  const float* adj  = (const float*)d_in[1];   // [N, N]
  const float* W    = (const float*)d_in[2];   // [IN_FT, OUT_FT]
  const float* cb   = (const float*)d_in[3];   // [OUT_FT]
  const float* eb   = (const float*)d_in[4];   // [OUT_FT]
  float* out = (float*)d_out;                  // [N, OUT_FT] fp32

  char* ws = (char*)d_ws;
  float*   dinv = (float*)ws;                          //  40 KB
  __bf16*  WT   = (__bf16*)(ws + 40960);               // 256 KB
  __bf16*  hsT  = (__bf16*)(ws + 40960 + 262144);      // 5.13 MB

  deg_kernel<<<NN, 256, 0, stream>>>(adj, dinv);
  wt_kernel<<<(INF * OUTF + 255) / 256, 256, 0, stream>>>(W, WT, hsT);
  h_kernel<<<NSTRIP, 256, 0, stream>>>(feat, WT, dinv, hsT);
  agg_kernel<<<(NSTRIP + 7) / 8, 256, 0, stream>>>(adj, hsT, dinv, cb, eb, out);

  (void)in_sizes; (void)n_in; (void)out_size; (void)ws_size;
}
